// MultiPhaseGRUModel_9912784519210
// MI455X (gfx1250) — compile-verified, matrix operand order checked by reading, not run on hardware
//
#include <hip/hip_runtime.h>
#include <hip/hip_bf16.h>
#include <math.h>

// ---------------------------------------------------------------------------
// MultiPhaseGRUModel fused kernel for gfx1250 (MI455X, wave32, WMMA)
//   B=512, T=512, F_IN=64, H=128, 3H=384
// ---------------------------------------------------------------------------

#define B_DIM   512
#define T_DIM   512
#define F_IN    64
#define H_DIM   128
#define G3      384           // 3*H
#define TILE_B  16            // batch rows per workgroup
#define NWAVES  8             // 256 threads

// padded LDS row strides (bf16 elements); all multiples of 8 (16B alignment)
// and NOT multiples of 128 (avoids 64-bank conflicts at 256B stride)
#define WHT_STRIDE 136        // WhT rows: n in [0,384), k in [0,128)
#define WXT_STRIDE 72         // WxT rows: n in [0,384), k in [0,64)
#define H_STRIDE   136        // h rows:   m in [0,16),  k in [0,128)
#define X_STRIDE   72         // x rows:   m in [0,16),  k in [0,64)

typedef __attribute__((ext_vector_type(16))) __bf16 v16bf;
typedef __attribute__((ext_vector_type(8)))  float  v8f;

static constexpr int    WHT_ELEMS  = G3 * WHT_STRIDE;          // 52224
static constexpr int    WXT_ELEMS  = G3 * WXT_STRIDE;          // 27648
static constexpr int    HB_ELEMS   = 2 * TILE_B * H_STRIDE;    // 4352
static constexpr int    XB_ELEMS   = 2 * TILE_B * X_STRIDE;    // 2304
static constexpr size_t SMEM_BYTES =
    (size_t)(WHT_ELEMS + WXT_ELEMS + HB_ELEMS + XB_ELEMS) * sizeof(__bf16); // 173056 B

__device__ __forceinline__ v8f wmma_bf16(v16bf a, v16bf b, v8f c) {
  // D = A(16x32 bf16) * B(32x16 bf16) + C(16x16 f32)
  return __builtin_amdgcn_wmma_f32_16x16x32_bf16(
      /*neg_a=*/false, a, /*neg_b=*/false, b,
      /*c_mod=*/(short)0, c, /*reuse_a=*/false, /*reuse_b=*/false);
}

// Load a 16x32 bf16 fragment (A operand, or B operand from a pre-transposed
// [n][k] LDS image) per ISA 7.12.2: lane L -> row (L&15), K-base 8*(L>>4);
// VGPR0..3 = K kb..kb+7, VGPR4..7 = K kb+16..kb+23.  Two ds_load_b128.
__device__ __forceinline__ v16bf load_frag(const __bf16* base, int row0,
                                           int stride, int k0, int lane) {
  const int r  = row0 + (lane & 15);
  const int kb = k0 + ((lane >> 4) << 3);
  const __bf16* p = base + r * stride + kb;
  union { uint4 u[2]; v16bf v; } f;
  f.u[0] = *(const uint4*)(p);
  f.u[1] = *(const uint4*)(p + 16);
  return f.v;
}

__device__ __forceinline__ float sigmoid_f(float x) {
  return 1.0f / (1.0f + __expf(-x));
}
__device__ __forceinline__ float tanh_f(float x) {
  // tanh via exp2-backed __expf (v_exp_f32)
  float e = __expf(2.0f * x);
  return (e - 1.0f) / (e + 1.0f);
}

// ---------------------------------------------------------------------------
// Kernel 1: fused input-projection + GRU scan. One WG per 16 batch rows.
// Weights resident in LDS (bf16, pre-transposed) for the entire T loop.
// ---------------------------------------------------------------------------
__global__ __launch_bounds__(256) void gru_scan_kernel(
    const float* __restrict__ inputs,   // (B, T, 65)
    const float* __restrict__ Wx,       // (64, 384)
    const float* __restrict__ Wh,       // (128, 384)
    const float* __restrict__ bias,     // (2, 384)
    float* __restrict__ hid)            // (B, 128) final hidden (workspace)
{
  extern __shared__ __align__(16) char smem_raw[];
  __bf16* WHT  = (__bf16*)smem_raw;          // [384][WHT_STRIDE]  WhT[n][k]
  __bf16* WXT  = WHT + WHT_ELEMS;            // [384][WXT_STRIDE]  WxT[n][k]
  __bf16* HBUF = WXT + WXT_ELEMS;            // [2][16][H_STRIDE]
  __bf16* XBUF = HBUF + HB_ELEMS;            // [2][16][X_STRIDE]

  const int tid  = threadIdx.x;
  const int lane = tid & 31;
  const int wave = tid >> 5;                 // 0..7 == H-column tile
  const int b0   = blockIdx.x * TILE_B;

  // ---- one-time weight preload: transpose + f32->bf16 --------------------
  for (int i = tid; i < H_DIM * G3; i += 256) {        // Wh[k][n] -> WHT[n][k]
    int k = i / G3, n = i - k * G3;
    WHT[n * WHT_STRIDE + k] = (__bf16)Wh[i];
  }
  for (int i = tid; i < F_IN * G3; i += 256) {         // Wx[k][n] -> WXT[n][k]
    int k = i / G3, n = i - k * G3;
    WXT[n * WXT_STRIDE + k] = (__bf16)Wx[i];
  }
  for (int i = tid; i < HB_ELEMS; i += 256) HBUF[i] = (__bf16)0.0f;  // h0 = 0

  // ---- per-lane loop invariants ------------------------------------------
  const int n16  = lane & 15;
  const int col  = wave * 16 + n16;          // H column this lane owns in C/D
  const float bz  = bias[col]            + bias[G3 + col];              // fused
  const float br  = bias[H_DIM + col]    + bias[G3 + H_DIM + col];      // fused
  const float bxh = bias[2 * H_DIM + col];                               // b_in_h
  const float brh = bias[G3 + 2 * H_DIM + col];                          // b_rec_h
  const int mbase = (lane >> 4) << 3;        // C/D: VGPR v -> row m = v + mbase

  const int xrow = tid >> 4;                 // x staging: row 0..15
  const int xcol = (tid & 15) << 2;          //            cols c..c+3

  __syncthreads();

  // ---- sequential scan over T --------------------------------------------
  for (int t = 0; t < T_DIM; ++t) {
    const int p = t & 1;

    // stage x[:, t, :64] (f32 global, row stride 65) -> bf16 LDS tile
    {
      const float* src =
          inputs + (((size_t)(b0 + xrow)) * T_DIM + t) * (F_IN + 1) + xcol;
      __bf16* dst = XBUF + p * TILE_B * X_STRIDE + xrow * X_STRIDE + xcol;
      dst[0] = (__bf16)src[0];
      dst[1] = (__bf16)src[1];
      dst[2] = (__bf16)src[2];
      dst[3] = (__bf16)src[3];
    }
    __syncthreads();   // x[t] staged; h[t] (written last iter) visible

    const __bf16* hsrc = HBUF + p * TILE_B * H_STRIDE;
    const __bf16* xsrc = XBUF + p * TILE_B * X_STRIDE;

    v8f accZ  = {0.f, 0.f, 0.f, 0.f, 0.f, 0.f, 0.f, 0.f};
    v8f accR  = {0.f, 0.f, 0.f, 0.f, 0.f, 0.f, 0.f, 0.f};
    v8f accXH = {0.f, 0.f, 0.f, 0.f, 0.f, 0.f, 0.f, 0.f};
    v8f accRH = {0.f, 0.f, 0.f, 0.f, 0.f, 0.f, 0.f, 0.f};

    // input projection: x(16x64) @ Wx(64x384), K = 2 tiles of 32
#pragma unroll
    for (int kt = 0; kt < 2; ++kt) {
      v16bf a  = load_frag(xsrc, 0, X_STRIDE, kt * 32, lane);
      v16bf bZ = load_frag(WXT, 0 * H_DIM + wave * 16, WXT_STRIDE, kt * 32, lane);
      accZ  = wmma_bf16(a, bZ, accZ);
      v16bf bR = load_frag(WXT, 1 * H_DIM + wave * 16, WXT_STRIDE, kt * 32, lane);
      accR  = wmma_bf16(a, bR, accR);
      v16bf bH = load_frag(WXT, 2 * H_DIM + wave * 16, WXT_STRIDE, kt * 32, lane);
      accXH = wmma_bf16(a, bH, accXH);
    }
    // recurrence: h(16x128) @ Wh(128x384), K = 4 tiles of 32
#pragma unroll
    for (int kt = 0; kt < 4; ++kt) {
      v16bf a  = load_frag(hsrc, 0, H_STRIDE, kt * 32, lane);
      v16bf bZ = load_frag(WHT, 0 * H_DIM + wave * 16, WHT_STRIDE, kt * 32, lane);
      accZ  = wmma_bf16(a, bZ, accZ);
      v16bf bR = load_frag(WHT, 1 * H_DIM + wave * 16, WHT_STRIDE, kt * 32, lane);
      accR  = wmma_bf16(a, bR, accR);
      v16bf bH = load_frag(WHT, 2 * H_DIM + wave * 16, WHT_STRIDE, kt * 32, lane);
      accRH = wmma_bf16(a, bH, accRH);
    }

    // gates + state update; write h[t+1] into the other buffer
    __bf16* hdst = HBUF + (1 - p) * TILE_B * H_STRIDE;
#pragma unroll
    for (int v = 0; v < 8; ++v) {
      const int m = v + mbase;               // C/D layout row
      float z  = sigmoid_f(accZ[v] + bz);
      float r  = sigmoid_f(accR[v] + br);
      float hh = tanh_f((accXH[v] + bxh) + r * (accRH[v] + brh));
      float ho = (float)hsrc[m * H_STRIDE + col];
      float hn = z * ho + (1.0f - z) * hh;
      hdst[m * H_STRIDE + col] = (__bf16)hn;
    }
    // (single barrier per step at top of loop; buffers have reuse distance 2)
  }

  __syncthreads();
  // final hidden lives in buffer 0 (t=511 wrote 1-(511&1)=0)
  for (int i = tid; i < TILE_B * H_DIM; i += 256) {
    int r = i >> 7, c = i & (H_DIM - 1);
    hid[(size_t)(b0 + r) * H_DIM + c] = (float)HBUF[r * H_STRIDE + c];
  }
}

// ---------------------------------------------------------------------------
// Kernel 2: tiny head.  hidden += W_state[idx]; relu(h@w1+b1); BN; @w2+b2
// One 64-thread block per batch row.
// ---------------------------------------------------------------------------
__global__ __launch_bounds__(64) void gru_head_kernel(
    const float* __restrict__ inputs, const float* __restrict__ hid,
    const float* __restrict__ W_state, const float* __restrict__ w1,
    const float* __restrict__ b1, const float* __restrict__ gamma,
    const float* __restrict__ beta, const float* __restrict__ mean,
    const float* __restrict__ var, const float* __restrict__ w2,
    const float* __restrict__ b2, float* __restrict__ out)
{
  __shared__ float hsh[H_DIM];
  __shared__ float red[64];
  const int b = blockIdx.x;
  const int j = threadIdx.x;

  // idx = clip(int(state[b, T-1]), 0, 2)
  float sv = inputs[((size_t)b * T_DIM + (T_DIM - 1)) * (F_IN + 1) + F_IN];
  int idx = (int)sv;
  idx = idx < 0 ? 0 : (idx > 2 ? 2 : idx);

  for (int k = j; k < H_DIM; k += 64)
    hsh[k] = hid[(size_t)b * H_DIM + k] + W_state[idx * H_DIM + k];
  __syncthreads();

  float acc = b1[j];
#pragma unroll 8
  for (int k = 0; k < H_DIM; ++k) acc = fmaf(hsh[k], w1[k * 64 + j], acc);
  float h1 = fmaxf(acc, 0.0f);
  float bn = (h1 - mean[j]) * rsqrtf(var[j] + 1e-3f) * gamma[j] + beta[j];
  red[j] = bn * w2[j];
  __syncthreads();
#pragma unroll
  for (int s = 32; s > 0; s >>= 1) {
    if (j < s) red[j] += red[j + s];
    __syncthreads();
  }
  if (j == 0) out[b] = red[0] + b2[0];
}

// ---------------------------------------------------------------------------
extern "C" void kernel_launch(void* const* d_in, const int* in_sizes, int n_in,
                              void* d_out, int out_size, void* d_ws, size_t ws_size,
                              hipStream_t stream) {
  const float* inputs  = (const float*)d_in[0];
  const float* Wx      = (const float*)d_in[1];
  const float* Wh      = (const float*)d_in[2];
  const float* bias    = (const float*)d_in[3];
  const float* W_state = (const float*)d_in[4];
  const float* w1      = (const float*)d_in[5];
  const float* b1      = (const float*)d_in[6];
  const float* gamma   = (const float*)d_in[7];
  const float* beta    = (const float*)d_in[8];
  const float* mean    = (const float*)d_in[9];
  const float* var     = (const float*)d_in[10];
  const float* w2      = (const float*)d_in[11];
  const float* b2      = (const float*)d_in[12];
  float* out = (float*)d_out;
  float* hid = (float*)d_ws;     // B*H*4 = 256 KB scratch

  // allow 173 KB dynamic LDS (CDNA5 WGP has 320 KB); no-op if already set
  hipFuncSetAttribute(reinterpret_cast<const void*>(gru_scan_kernel),
                      hipFuncAttributeMaxDynamicSharedMemorySize,
                      (int)SMEM_BYTES);

  gru_scan_kernel<<<dim3(B_DIM / TILE_B), dim3(256), SMEM_BYTES, stream>>>(
      inputs, Wx, Wh, bias, hid);
  gru_head_kernel<<<dim3(B_DIM), dim3(64), 0, stream>>>(
      inputs, hid, W_state, w1, b1, gamma, beta, mean, var, w2, b2, out);
}
